// DenseDilatedKnnGraph_43439299232096
// MI455X (gfx1250) — matrix operand order repaired and checked by reading, not compiled, not run on hardware
//
#include <hip/hip_runtime.h>

typedef __attribute__((ext_vector_type(16))) _Float16 v16h;
typedef __attribute__((ext_vector_type(8)))  float    v8f;
typedef __attribute__((ext_vector_type(4)))  unsigned int u32x4;

#define BATCH 4
#define NPTS  8192
#define CDIM  64
#define ROWS  128     // query rows per workgroup (one per thread for selection)
#define CTILE 64      // key columns streamed per iteration
#define NTILES (NPTS / CTILE)
#define LSTRIDE (CTILE + 1)   // LDS row stride (65) -> bank-conflict free
#define KK    18      // K * DILATION
#define KOUT  9

// ---------------------------------------------------------------------------
// Kernel 1: L2-normalize each point over C=64 and emit f16 points row-major
// [b*N + n][c].  Reads of x are coalesced (adjacent threads = adjacent n for
// fixed c).  Whole tensor is 8 MB -> L2 resident for the second pass.
// ---------------------------------------------------------------------------
__global__ __launch_bounds__(256) void normalize_f16_kernel(
    const float* __restrict__ x, _Float16* __restrict__ pts)
{
    int p = blockIdx.x * blockDim.x + threadIdx.x;   // 0 .. B*N-1
    int b = p >> 13;                                 // / 8192
    int n = p & (NPTS - 1);
    const float* xb = x + (size_t)b * CDIM * NPTS + n;

    float s = 0.f;
#pragma unroll
    for (int c = 0; c < CDIM; ++c) {
        float v = xb[(size_t)c * NPTS];
        s += v * v;
    }
    float scale = 1.0f / fmaxf(sqrtf(s), 1e-12f);

    union { _Float16 h[CDIM]; u32x4 q[CDIM / 8]; } u;
#pragma unroll
    for (int c = 0; c < CDIM; ++c)
        u.h[c] = (_Float16)(xb[(size_t)c * NPTS] * scale);

    u32x4* dst = (u32x4*)(pts + (size_t)p * CDIM);
#pragma unroll
    for (int i = 0; i < CDIM / 8; ++i)
        dst[i] = u.q[i];
}

// ---------------------------------------------------------------------------
// Kernel 2: fused Gram-tile WMMA + streaming per-row top-18, double-buffered
// so selection VALU of tile t-1 co-executes with WMMA of tile t.
// Grid: BATCH * (NPTS/ROWS) = 256 blocks of 128 threads (4 waves, wave32).
// Wave w computes rows [32w, 32w+32) of the block's 128 rows.
// ---------------------------------------------------------------------------
__global__ __launch_bounds__(128) void knn_wmma_kernel(
    const _Float16* __restrict__ pts, int* __restrict__ out)
{
    __shared__ float sims[2][ROWS * LSTRIDE];    // 2 x 32.5 KB of the 320 KB WGP LDS

    const int wid  = threadIdx.x >> 5;
    const int lane = threadIdx.x & 31;
    const int l16  = lane & 15;
    const int hi   = lane >> 4;          // 0: lanes 0-15, 1: lanes 16-31

    const int b  = blockIdx.x >> 6;              // NPTS/ROWS = 64 blocks/batch
    const int rb = (blockIdx.x & 63) * ROWS;

    // each point = 64 halves = 8 x uint4 (128 B)
    const u32x4* base = (const u32x4*)(pts + (size_t)b * NPTS * CDIM);

    // --- A fragments (registers for the whole kernel) ----------------------
    // ISA 16-bit A 16x32 layout: lane<16 holds row M=lane, halves K 0-7 &
    // 16-23; lane>=16 holds row M=lane-16, halves K 8-15 & 24-31.
    v16h A[2][2];
#pragma unroll
    for (int s2 = 0; s2 < 2; ++s2) {
        int row = rb + wid * 32 + s2 * 16 + l16;
        int c0  = hi * 8;                          // half offset 0 or 8
#pragma unroll
        for (int kb = 0; kb < 2; ++kb) {
            union { v16h h; u32x4 q[2]; } u;
            int hb = kb * 32 + c0;
            u.q[0] = base[row * 8 + (hb >> 3)];
            u.q[1] = base[row * 8 + ((hb + 16) >> 3)];
            A[s2][kb] = u.h;
        }
    }

    // --- streaming top-18, kept sorted descending by similarity ------------
    float bestv[KK];
    int   besti[KK];
#pragma unroll
    for (int i = 0; i < KK; ++i) { bestv[i] = -3.0e38f; besti[i] = 0; }

    v8f acc[2][4];   // [row strip][col subtile] accumulators for one tile

    // issue B loads + 16 chained WMMAs for column tile at ct -> acc[][]
    auto compute_tile = [&](int ct) {
        if (ct + CTILE < NPTS) {   // prefetch next key tile while computing
            const char* pf = (const char*)base + (size_t)(ct + CTILE + lane * 2) * 128;
            __builtin_prefetch(pf, 0, 3);
        }
#pragma unroll
        for (int sub = 0; sub < 4; ++sub) {
            int col = ct + sub * 16 + l16;
            int kof = hi * 8;                // B: lane<16 -> K 0-7 & 16-23, else 8-15 & 24-31
            v16h B0, B1;
            { union { v16h h; u32x4 q[2]; } u;
              u.q[0] = base[col * 8 + (kof >> 3)];
              u.q[1] = base[col * 8 + ((kof + 16) >> 3)];
              B0 = u.h; }
            { union { v16h h; u32x4 q[2]; } u;
              u.q[0] = base[col * 8 + ((32 + kof) >> 3)];
              u.q[1] = base[col * 8 + ((48 + kof) >> 3)];
              B1 = u.h; }
#pragma unroll
            for (int s2 = 0; s2 < 2; ++s2) {
                v8f c = {};
                c = __builtin_amdgcn_wmma_f32_16x16x32_f16(
                        false, A[s2][0], false, B0, (short)0, c, false, false);
                c = __builtin_amdgcn_wmma_f32_16x16x32_f16(
                        false, A[s2][1], false, B1, (short)0, c, false, false);
                acc[s2][sub] = c;
            }
        }
    };

    // drain acc[][] into LDS buffer `buf`
    // C/D layout: lanes 0-15 -> M=v, lanes 16-31 -> M=v+8; N = lane%16
    auto store_tile = [&](int buf) {
#pragma unroll
        for (int sub = 0; sub < 4; ++sub) {
            int cloc = sub * 16 + l16;
#pragma unroll
            for (int s2 = 0; s2 < 2; ++s2) {
                int rloc = wid * 32 + s2 * 16 + hi * 8;
#pragma unroll
                for (int v = 0; v < 8; ++v)
                    sims[buf][(rloc + v) * LSTRIDE + cloc] = acc[s2][sub][v];
            }
        }
    };

    // top-18 sorted insert over LDS buffer `buf` holding columns [ct, ct+64)
    auto select_tile = [&](int buf, int ct) {
        const float* srow = &sims[buf][threadIdx.x * LSTRIDE];
#pragma unroll 4
        for (int j = 0; j < CTILE; ++j) {
            float s = srow[j];
            if (s > bestv[KK - 1]) {               // rare path after warm-up
                float cv = s; int cid = ct + j;
#pragma unroll
                for (int i = 0; i < KK; ++i) {
                    if (cv > bestv[i]) {           // strict: stable tie order
                        float tv = bestv[i]; int ti = besti[i];
                        bestv[i] = cv; besti[i] = cid;
                        cv = tv; cid = ti;
                    }
                }
            }
        }
    };

    // --- software pipeline: one barrier per tile ---------------------------
    compute_tile(0);
    store_tile(0);
    __syncthreads();
    int buf = 0;
    for (int t = 1; t < NTILES; ++t) {
        compute_tile(t * CTILE);             // WMMAs in flight...
        select_tile(buf, (t - 1) * CTILE);   // ...while selection VALU runs
        store_tile(buf ^ 1);
        __syncthreads();
        buf ^= 1;
    }
    select_tile(buf, NPTS - CTILE);

    // --- output: edge_index (2, B, N, 9); dilation picks ranks 0,2,...,16 --
    int myrow = rb + threadIdx.x;
    size_t o0 = ((size_t)b * NPTS + myrow) * KOUT;
    size_t o1 = (size_t)BATCH * NPTS * KOUT + o0;
#pragma unroll
    for (int j = 0; j < KOUT; ++j) out[o0 + j] = besti[2 * j];
#pragma unroll
    for (int j = 0; j < KOUT; ++j) out[o1 + j] = myrow;
}

extern "C" void kernel_launch(void* const* d_in, const int* in_sizes, int n_in,
                              void* d_out, int out_size, void* d_ws, size_t ws_size,
                              hipStream_t stream)
{
    const float* x   = (const float*)d_in[0];        // (4, 64, 8192, 1) f32
    _Float16*    pts = (_Float16*)d_ws;              // 4 MB f16 points, row-major
    int*         o   = (int*)d_out;                  // (2, 4, 8192, 9) int32

    normalize_f16_kernel<<<(BATCH * NPTS) / 256, 256, 0, stream>>>(x, pts);
    knn_wmma_kernel<<<BATCH * (NPTS / ROWS), ROWS, 0, stream>>>(pts, o);
}
